// DenseAttention_13743895347586
// MI455X (gfx1250) — compile-verified, matrix operand order checked by reading, not run on hardware
//
#include <hip/hip_runtime.h>
#include <hip/hip_bf16.h>

// ---------------------------------------------------------------------------
// Problem constants
// ---------------------------------------------------------------------------
constexpr int Ccls = 31;     // classes
constexpr int Bn   = 1024;   // batch
constexpr int Dn   = 1024;   // feature dim
constexpr int HALFB = Bn / 2;

// GEMM tiling
constexpr int BM = 128, BN = 128, BK = 32;
constexpr int LDST = 40;     // LDS row stride in bf16 (80B = 20 dwords, conflict-free frags)

typedef __attribute__((ext_vector_type(8)))  __bf16 v8bf;
typedef __attribute__((ext_vector_type(16))) __bf16 v16bf;
typedef __attribute__((ext_vector_type(8)))  float  v8f;

// fp32 -> bf16 round-to-nearest-even (bit form + packers)
static __device__ inline unsigned bfbits(float f) {
  unsigned u = __float_as_uint(f);
  return (u + 0x7FFFu + ((u >> 16) & 1u)) >> 16;
}
static __device__ inline unsigned pack2bf(float lo, float hi) {
  return bfbits(lo) | (bfbits(hi) << 16);
}
static __device__ inline __bf16 f2bf(float f) {
  unsigned short h = (unsigned short)bfbits(f);
  return __builtin_bit_cast(__bf16, h);
}

// Gather one WMMA operand fragment from an LDS tile stored [row][k] with
// stride LDST. Per ISA 16-bit A/B layout: lane l holds row (l&15); elements
// 0..7 = K[base..base+7], elements 8..15 = K[base+16..base+23], base=(l>>4)*8.
static __device__ inline v16bf load_frag(const __bf16* p) {
  v8bf lo = *(const v8bf*)(p);
  v8bf hi = *(const v8bf*)(p + 16);
  return __builtin_shufflevector(lo, hi, 0,1,2,3,4,5,6,7,8,9,10,11,12,13,14,15);
}

// ---------------------------------------------------------------------------
// GEMM1: xB[c] = X @ K[c] + bias[c], output bf16 [C,B,D]
// ---------------------------------------------------------------------------
__global__ __launch_bounds__(256) void gemm1_k(const float* __restrict__ X,
                                               const float* __restrict__ Kw,
                                               const float* __restrict__ bias,
                                               __bf16* __restrict__ xB)
{
  __shared__ __bf16 sA[BM * LDST];
  __shared__ __bf16 sB[BN * LDST];

  const int c    = blockIdx.z;
  const int m0   = blockIdx.y * BM;
  const int n0   = blockIdx.x * BN;
  const int t    = threadIdx.x;
  const int lane = t & 31, wave = t >> 5;
  const int wm   = wave & 3, wn = wave >> 2;       // 4x2 wave grid
  const int half = lane >> 4, r16 = lane & 15;

  const float* Kc = Kw + (size_t)c * Dn * Dn;

  const v8f vzero = {0.f,0.f,0.f,0.f,0.f,0.f,0.f,0.f};
  v8f acc[2][4];
  #pragma unroll
  for (int mi = 0; mi < 2; ++mi)
    #pragma unroll
    for (int ni = 0; ni < 4; ++ni) acc[mi][ni] = vzero;

  for (int k0 = 0; k0 < Dn; k0 += BK) {
    // ---- Stage A (BMxBK): batch all global loads first, then cvt+store ----
    float4 va[4];
    #pragma unroll
    for (int e = 0; e < 4; ++e) {
      int u = t + e * 256;
      int rr = u >> 3, qq = u & 7;
      va[e] = *(const float4*)(X + (size_t)(m0 + rr) * Dn + k0 + qq * 4);
    }
    // ---- Stage B transposed: sB[n][k] = Kc[k0+k][n0+n]; batch loads -------
    float fb0[8], fb1[8];
    #pragma unroll
    for (int e = 0; e < 8; ++e) {
      int u = t + e * 256;
      int np = u & 127, kp = u >> 7;                 // kp = k-pair index 0..15
      const float* g = Kc + (size_t)(k0 + 2 * kp) * Dn + n0 + np;
      fb0[e] = g[0];
      fb1[e] = g[Dn];
    }
    if (k0 + BK < Dn) {   // hint next K-tile into cache (global_prefetch_b8)
      __builtin_prefetch(X  + (size_t)(m0 + (t >> 1)) * Dn + k0 + BK + (t & 1) * 16, 0, 1);
      __builtin_prefetch(Kc + (size_t)(k0 + BK + (t >> 3)) * Dn + n0 + (t & 7) * 16, 0, 1);
    }
    #pragma unroll
    for (int e = 0; e < 4; ++e) {
      int u = t + e * 256;
      int rr = u >> 3, qq = u & 7;
      uint2 pk;
      pk.x = pack2bf(va[e].x, va[e].y);
      pk.y = pack2bf(va[e].z, va[e].w);
      *(uint2*)(sA + rr * LDST + qq * 4) = pk;       // ds_store_b64
    }
    #pragma unroll
    for (int e = 0; e < 8; ++e) {
      int u = t + e * 256;
      int np = u & 127, kp = u >> 7;
      *(unsigned*)(sB + np * LDST + 2 * kp) = pack2bf(fb0[e], fb1[e]); // ds_store_b32
    }
    __syncthreads();

    v16bf aF[2], bF[4];
    #pragma unroll
    for (int mi = 0; mi < 2; ++mi)
      aF[mi] = load_frag(sA + (wm * 32 + mi * 16 + r16) * LDST + half * 8);
    #pragma unroll
    for (int ni = 0; ni < 4; ++ni)
      bF[ni] = load_frag(sB + (wn * 64 + ni * 16 + r16) * LDST + half * 8);
    #pragma unroll
    for (int mi = 0; mi < 2; ++mi)
      #pragma unroll
      for (int ni = 0; ni < 4; ++ni)
        acc[mi][ni] = __builtin_amdgcn_wmma_f32_16x16x32_bf16(
            false, aF[mi], false, bF[ni], (short)0, acc[mi][ni], false, false);
    __syncthreads();
  }

  // Epilogue: add bias, convert to bf16, store.
  // C/D layout: N = lane&15, M = (lane>>4)*8 + vgpr.
  __bf16* xBc = xB + (size_t)c * Bn * Dn;
  #pragma unroll
  for (int mi = 0; mi < 2; ++mi)
    #pragma unroll
    for (int ni = 0; ni < 4; ++ni) {
      int n = n0 + wn * 64 + ni * 16 + r16;
      float bv = bias[(size_t)c * Dn + n];
      #pragma unroll
      for (int v = 0; v < 8; ++v) {
        int i = m0 + wm * 32 + mi * 16 + half * 8 + v;
        xBc[(size_t)i * Dn + n] = f2bf(acc[mi][ni][v] + bv);
      }
    }
}

// ---------------------------------------------------------------------------
// GEMM2: scores = xB[c] @ xB[c]^T, masked, stored (i,j,c)-flat (softmax order)
// Staging uses the CDNA5 async global->LDS DMA path (ASYNCcnt).
// ---------------------------------------------------------------------------
__global__ __launch_bounds__(256) void gemm2_k(const __bf16* __restrict__ xB,
                                               const int* __restrict__ labels,
                                               float* __restrict__ S)
{
  __shared__ __bf16 sA[BM * LDST];
  __shared__ __bf16 sB[BN * LDST];
  __shared__ int sLabR[BM];
  __shared__ int sLabC[BN];

  const int c    = blockIdx.z;
  const int m0   = blockIdx.y * BM;
  const int n0   = blockIdx.x * BN;
  const int t    = threadIdx.x;
  const int lane = t & 31, wave = t >> 5;
  const int wm   = wave & 3, wn = wave >> 2;
  const int half = lane >> 4, r16 = lane & 15;

  const __bf16* xBc = xB + (size_t)c * Bn * Dn;

  if (t < BM)       sLabR[t]       = labels[(size_t)(m0 + t) * Ccls + c];
  else              sLabC[t - BM]  = labels[(size_t)(n0 + t - BM) * Ccls + c];

  const v8f vzero = {0.f,0.f,0.f,0.f,0.f,0.f,0.f,0.f};
  v8f acc[2][4];
  #pragma unroll
  for (int mi = 0; mi < 2; ++mi)
    #pragma unroll
    for (int ni = 0; ni < 4; ++ni) acc[mi][ni] = vzero;

  for (int k0 = 0; k0 < Dn; k0 += BK) {
    // Async DMA both operand tiles straight into LDS (no VGPR round-trip).
    // Per-lane 16B transfers; LDS dest = low 32 bits of the shared pointer.
    #pragma unroll
    for (int u = t; u < BM * BK / 8; u += 256) {   // 16B chunks
      int rr = u >> 2, qq = u & 3;
      unsigned ldst = (unsigned)(size_t)(sA + rr * LDST + qq * 8);
      const __bf16* g = xBc + (size_t)(m0 + rr) * Dn + k0 + qq * 8;
      asm volatile("global_load_async_to_lds_b128 %0, %1, off"
                   :: "v"(ldst), "v"(g) : "memory");
    }
    #pragma unroll
    for (int u = t; u < BN * BK / 8; u += 256) {
      int rr = u >> 2, qq = u & 3;
      unsigned ldst = (unsigned)(size_t)(sB + rr * LDST + qq * 8);
      const __bf16* g = xBc + (size_t)(n0 + rr) * Dn + k0 + qq * 8;
      asm volatile("global_load_async_to_lds_b128 %0, %1, off"
                   :: "v"(ldst), "v"(g) : "memory");
    }
    asm volatile("s_wait_asynccnt 0x0" ::: "memory");
    __syncthreads();

    v16bf aF[2], bF[4];
    #pragma unroll
    for (int mi = 0; mi < 2; ++mi)
      aF[mi] = load_frag(sA + (wm * 32 + mi * 16 + r16) * LDST + half * 8);
    #pragma unroll
    for (int ni = 0; ni < 4; ++ni)
      bF[ni] = load_frag(sB + (wn * 64 + ni * 16 + r16) * LDST + half * 8);
    #pragma unroll
    for (int mi = 0; mi < 2; ++mi)
      #pragma unroll
      for (int ni = 0; ni < 4; ++ni)
        acc[mi][ni] = __builtin_amdgcn_wmma_f32_16x16x32_bf16(
            false, aF[mi], false, bF[ni], (short)0, acc[mi][ni], false, false);
    __syncthreads();
  }

  // Epilogue: mask (label equality AND cross-domain), write masked score in
  // (i,j,c)-flat order so softmax groups (flat>>20) are contiguous.
  #pragma unroll
  for (int mi = 0; mi < 2; ++mi)
    #pragma unroll
    for (int ni = 0; ni < 4; ++ni) {
      int jl = wn * 64 + ni * 16 + r16;
      int j  = n0 + jl;
      int lc = sLabC[jl];
      bool jlo = (j < HALFB);
      #pragma unroll
      for (int v = 0; v < 8; ++v) {
        int il = wm * 32 + mi * 16 + half * 8 + v;
        int i  = m0 + il;
        bool kept = (sLabR[il] == lc) && ((i < HALFB) != jlo);
        float s = kept ? acc[mi][ni][v] : 0.f;
        S[(size_t)(i * Bn + j) * Ccls + c] = s;
      }
    }
}

// ---------------------------------------------------------------------------
// Deterministic softmax statistics over the 31 flat groups of 2^20 elements.
// ---------------------------------------------------------------------------
constexpr int CHUNK = 8192;          // per block; 128 blocks per group
constexpr int NCHUNK = (1 << 20) / CHUNK;

__global__ __launch_bounds__(256) void pmax_k(const float* __restrict__ S,
                                              float* __restrict__ pmax) {
  int g = blockIdx.y, b = blockIdx.x;
  size_t base = ((size_t)g << 20) + (size_t)b * CHUNK + threadIdx.x;
  float m = 0.f;  // every group contains masked zeros, so group max >= 0
  #pragma unroll
  for (int e = 0; e < CHUNK / 256; ++e) m = fmaxf(m, S[base + e * 256]);
  __shared__ float red[256];
  red[threadIdx.x] = m; __syncthreads();
  for (int s = 128; s; s >>= 1) {
    if (threadIdx.x < (unsigned)s) red[threadIdx.x] = fmaxf(red[threadIdx.x], red[threadIdx.x + s]);
    __syncthreads();
  }
  if (threadIdx.x == 0) pmax[g * NCHUNK + b] = red[0];
}

__global__ __launch_bounds__(128) void rmax_k(const float* __restrict__ pmax,
                                              float* __restrict__ maxbuf) {
  __shared__ float red[128];
  int g = blockIdx.x;
  red[threadIdx.x] = pmax[g * NCHUNK + threadIdx.x]; __syncthreads();
  for (int s = 64; s; s >>= 1) {
    if (threadIdx.x < (unsigned)s) red[threadIdx.x] = fmaxf(red[threadIdx.x], red[threadIdx.x + s]);
    __syncthreads();
  }
  if (threadIdx.x == 0) maxbuf[g] = red[0];
}

__global__ __launch_bounds__(256) void psum_k(const float* __restrict__ S,
                                              const float* __restrict__ maxbuf,
                                              float* __restrict__ psum) {
  int g = blockIdx.y, b = blockIdx.x;
  float m = maxbuf[g];
  size_t base = ((size_t)g << 20) + (size_t)b * CHUNK + threadIdx.x;
  float p = 0.f;   // masked entries stored as 0 contribute exp(0-m): faithful
  #pragma unroll
  for (int e = 0; e < CHUNK / 256; ++e) p += __expf(S[base + e * 256] - m);
  __shared__ float red[256];
  red[threadIdx.x] = p; __syncthreads();
  for (int s = 128; s; s >>= 1) {
    if (threadIdx.x < (unsigned)s) red[threadIdx.x] += red[threadIdx.x + s];
    __syncthreads();
  }
  if (threadIdx.x == 0) psum[g * NCHUNK + b] = red[0];
}

__global__ __launch_bounds__(128) void rsum_k(const float* __restrict__ psum,
                                              float* __restrict__ denom) {
  __shared__ float red[128];
  int g = blockIdx.x;
  red[threadIdx.x] = psum[g * NCHUNK + threadIdx.x]; __syncthreads();
  for (int s = 64; s; s >>= 1) {
    if (threadIdx.x < (unsigned)s) red[threadIdx.x] += red[threadIdx.x + s];
    __syncthreads();
  }
  if (threadIdx.x == 0) denom[g] = red[0];
}

// ---------------------------------------------------------------------------
// Final: out[i,j] = sum_c kept(i,j,c) * exp(S - m_g) / denom_g
// ---------------------------------------------------------------------------
__global__ __launch_bounds__(256) void final_k(const float* __restrict__ S,
                                               const int* __restrict__ labels,
                                               const float* __restrict__ maxbuf,
                                               const float* __restrict__ denom,
                                               float* __restrict__ out) {
  __shared__ float sm[Ccls], sdi[Ccls];
  int t = threadIdx.x;
  if (t < Ccls) { sm[t] = maxbuf[t]; sdi[t] = 1.0f / denom[t]; }
  __syncthreads();
  int idx = blockIdx.x * 256 + t;
  int i = idx >> 10, j = idx & (Bn - 1);
  float r = 0.f;
  if ((i < HALFB) != (j < HALFB)) {
    size_t fb = (size_t)idx * Ccls;
    #pragma unroll 1
    for (int c = 0; c < Ccls; ++c) {
      if (labels[(size_t)i * Ccls + c] == labels[(size_t)j * Ccls + c]) {
        int g = (int)((fb + c) >> 20);
        r += __expf(S[fb + c] - sm[g]) * sdi[g];
      }
    }
  }
  out[idx] = r;
}

// ---------------------------------------------------------------------------
// Launch
// ---------------------------------------------------------------------------
extern "C" void kernel_launch(void* const* d_in, const int* in_sizes, int n_in,
                              void* d_out, int out_size, void* d_ws, size_t ws_size,
                              hipStream_t stream) {
  (void)in_sizes; (void)n_in; (void)out_size; (void)ws_size;
  const float* X      = (const float*)d_in[0];   // [B,D]
  const int*   labels = (const int*)  d_in[1];   // [B,C]
  const float* Kw     = (const float*)d_in[2];   // [C,D,D]
  const float* bias   = (const float*)d_in[3];   // [C,D]
  float* out = (float*)d_out;                    // [B,B]

  // Workspace layout (~186 MB total):
  //   xB   bf16 [C,B,D]              62.0 MB
  //   S    f32  flat (i,j,c)        124.0 MB  (fits in 192MB L2)
  //   pmax/psum f32 [31*128]          32 KB
  //   maxbuf/denom f32 [31]           <1 KB
  char* ws = (char*)d_ws;
  __bf16* xB   = (__bf16*)ws;
  float* S     = (float*)(ws + 65011712ull);
  float* pmax  = (float*)(ws + 195035136ull);
  float* psum  = (float*)(ws + 195051008ull);
  float* maxb  = (float*)(ws + 195066880ull);
  float* denom = (float*)(ws + 195067136ull);

  dim3 blk(256);
  dim3 g1(Dn / BN, Bn / BM, Ccls);
  gemm1_k<<<g1, blk, 0, stream>>>(X, Kw, bias, xB);

  dim3 g2(Bn / BN, Bn / BM, Ccls);
  gemm2_k<<<g2, blk, 0, stream>>>(xB, labels, S);

  dim3 gp(NCHUNK, Ccls);
  pmax_k<<<gp, blk, 0, stream>>>(S, pmax);
  rmax_k<<<Ccls, 128, 0, stream>>>(pmax, maxb);
  psum_k<<<gp, blk, 0, stream>>>(S, maxb, psum);
  rsum_k<<<Ccls, 128, 0, stream>>>(psum, denom);
  final_k<<<(Bn * Bn) / 256, blk, 0, stream>>>(S, labels, maxb, denom, out);
}
